// LatticeMultiHeadAttention_35467839930938
// MI455X (gfx1250) — matrix operand order repaired and checked by reading, not compiled, not run on hardware
//
#include <hip/hip_runtime.h>
#include <hip/hip_bf16.h>
#include <math.h>

// Problem constants (match reference)
#define BB  2
#define LL  2048
#define DD  1024
#define HH  16
#define DKK 64

typedef __attribute__((ext_vector_type(16))) __bf16       v16bf;
typedef __attribute__((ext_vector_type(4)))  __bf16       v4bf;
typedef __attribute__((ext_vector_type(8)))  float        v8f;
typedef __attribute__((ext_vector_type(4)))  float        v4f;
typedef __attribute__((ext_vector_type(4)))  unsigned int u32x4;

// ---------------------------------------------------------------------------
// CDNA5 async global->LDS copy (ASYNCcnt-tracked, no VGPR data movement).
// VDST = LDS byte address (low 32 bits of the flat shared pointer),
// ADDR = 64-bit global address, GV mode (saddr = off).
// ---------------------------------------------------------------------------
static __device__ inline unsigned lds_lo32(const void* p) {
  return (unsigned)(unsigned long long)p;
}
static __device__ inline void async_copy_b128(unsigned lds_off,
                                              const void* gptr) {
  asm volatile("global_load_async_to_lds_b128 %0, %1, off"
               :: "v"(lds_off), "v"(gptr) : "memory");
}
#define WAIT_ASYNC(n) asm volatile("s_wait_asynccnt " #n ::: "memory")

// ---------------------------------------------------------------------------
// WMMA operand loaders (CDNA5 16-bit layouts, wave32).
// A (16x32, MxK): lane l<16 holds row M=l, K in {0..7, 16..23};
//                 lane l>=16 holds row M=l-16, K in {8..15, 24..31}.
// B (32x16, KxN): lane n<16 holds col N=n, K in {0..7, 16..23};
//                 lane n>=16 holds col N=n-16, K in {8..15, 24..31}.
// Both patterns are 16-byte contiguous chunks -> b128 loads.
// ---------------------------------------------------------------------------
static __device__ inline v16bf load_mat_a(const __bf16* base, int ld) {
  const int lane = threadIdx.x & 31;
  const __bf16* row = base + (size_t)(lane & 15) * ld + ((lane >> 4) << 3);
  union { u32x4 q[2]; v16bf v; } r;
  r.q[0] = *(const u32x4*)(row);        // K = kb .. kb+7
  r.q[1] = *(const u32x4*)(row + 16);   // K = kb+16 .. kb+23
  return r.v;
}

static __device__ inline v16bf load_mat_b(const __bf16* base, int ld) {
  const int lane = threadIdx.x & 31;
  const __bf16* col = base + (size_t)(lane & 15) * ld + ((lane >> 4) << 3);
  union { u32x4 q[2]; v16bf v; } r;
  r.q[0] = *(const u32x4*)(col);        // K = kb .. kb+7
  r.q[1] = *(const u32x4*)(col + 16);   // K = kb+16 .. kb+23
  return r.v;
}

static __device__ inline v8f wmma_bf16(v16bf a, v16bf b, v8f c) {
  return __builtin_amdgcn_wmma_f32_16x16x32_bf16(
      /*neg_a=*/false, a, /*neg_b=*/false, b,
      /*c_mod=*/(short)0, c, /*reuse_a=*/false, /*reuse_b=*/false);
}

// ---------------------------------------------------------------------------
// Kernel 1: Kuramoto phase chain (H=16, strictly sequential) -> bias[h]
// ---------------------------------------------------------------------------
__global__ void phases_bias_kernel(const float* __restrict__ cs,
                                   const float* __restrict__ nbw,
                                   const int*   __restrict__ nbidx,
                                   float* __restrict__ bias) {
  if (threadIdx.x == 0 && blockIdx.x == 0) {
    float ph[HH];
    for (int i = 0; i < HH; ++i) ph[i] = 0.0f;
    const float two_pi = 6.283185307179586f;
    for (int i = 0; i < HH; ++i) {
      float coupling = 0.0f;
      for (int k = 0; k < 8; ++k) {
        int j = nbidx[i * 8 + k];
        j = j < 0 ? 0 : (j > HH - 1 ? HH - 1 : j);
        coupling += nbw[i * 8 + k] * sinf(ph[j] - ph[i]);
      }
      float freq = 1.0f + 0.1f * (float)i;
      float nv = ph[i] + 0.01f * (freq + cs[i] * coupling);
      nv = fmodf(nv, two_pi);
      if (nv < 0.0f) nv += two_pi;
      ph[i] = nv;
    }
    for (int i = 0; i < HH; ++i) bias[i] = 0.1f * cosf(ph[i]);
  }
}

// ---------------------------------------------------------------------------
// Kernel 2: fp32 -> bf16 (weights + activations; converted once, reused)
// ---------------------------------------------------------------------------
__global__ void f32_to_bf16_kernel(const float* __restrict__ in,
                                   __bf16* __restrict__ out, int n) {
  int i = (blockIdx.x * blockDim.x + threadIdx.x) * 4;
  if (i + 3 < n) {
    v4f v = *(const v4f*)(in + i);
    v4bf o;
    o[0] = (__bf16)v[0]; o[1] = (__bf16)v[1];
    o[2] = (__bf16)v[2]; o[3] = (__bf16)v[3];
    *(v4bf*)(out + i) = o;
  }
}

// ---------------------------------------------------------------------------
// Kernel 3: WMMA GEMM  C[M,N] = A[M,K] * Bt[N,K]^T  (+ optional bias)
// Block = 128 threads (4 waves). Block tile 128x64, K-step 32.
// Wave w computes rows [w*32, w*32+32) x 64 cols (8 accumulators).
// Double-buffered: async global->LDS copy of stage k+1 overlaps WMMA of
// stage k; s_wait_asynccnt 0x6 retires the older 6-per-thread issue group.
// ---------------------------------------------------------------------------
static __device__ inline void gemm_issue_stage(__bf16* sA, __bf16* sB,
                                               const __bf16* A,
                                               const __bf16* Bt,
                                               int m0, int n0, int k0, int K,
                                               int t) {
  // A tile: 128 rows x 32 cols bf16 = 512 x 16B chunks -> 4 per thread.
#pragma unroll
  for (int j = 0; j < 4; ++j) {
    int ci = t + j * 128;
    int r = ci >> 2, c = (ci & 3) * 8;
    async_copy_b128(lds_lo32(sA + r * 32 + c),
                    A + (size_t)(m0 + r) * K + k0 + c);
  }
  // B tile: 64 rows x 32 cols bf16 = 256 x 16B chunks -> 2 per thread.
#pragma unroll
  for (int j = 0; j < 2; ++j) {
    int ci = t + j * 128;
    int r = ci >> 2, c = (ci & 3) * 8;
    async_copy_b128(lds_lo32(sB + r * 32 + c),
                    Bt + (size_t)(n0 + r) * K + k0 + c);
  }
}

template <bool OUTF32>
__global__ void __launch_bounds__(128)
gemm_bt_wmma(const __bf16* __restrict__ A, const __bf16* __restrict__ Bt,
             void* __restrict__ Cp, const float* __restrict__ bias,
             int M, int N, int K) {
  __shared__ __align__(16) __bf16 sA[2][128 * 32];
  __shared__ __align__(16) __bf16 sB[2][64 * 32];
  const int t = threadIdx.x;
  const int lane = t & 31;
  const int w = t >> 5;
  const int m0 = blockIdx.y * 128;
  const int n0 = blockIdx.x * 64;

  v8f acc0[4], acc1[4];
  v8f z = {0.f, 0.f, 0.f, 0.f, 0.f, 0.f, 0.f, 0.f};
#pragma unroll
  for (int i = 0; i < 4; ++i) { acc0[i] = z; acc1[i] = z; }

  // Prologue: stage 0 in flight.
  gemm_issue_stage(sA[0], sB[0], A, Bt, m0, n0, 0, K, t);

  int s = 0;
  for (int k0 = 0; k0 < K; k0 += 32, s ^= 1) {
    if (k0 + 32 < K) {
      gemm_issue_stage(sA[s ^ 1], sB[s ^ 1], A, Bt, m0, n0, k0 + 32, K, t);
      WAIT_ASYNC(0x6);   // retire stage k (6 issues/thread), keep k+1 flying
    } else {
      WAIT_ASYNC(0x0);   // last stage: drain everything
    }
    __syncthreads();

    v16bf a0 = load_mat_a(sA[s] + (w * 32) * 32, 32);
    v16bf a1 = load_mat_a(sA[s] + (w * 32 + 16) * 32, 32);
#pragma unroll
    for (int nt = 0; nt < 4; ++nt) {
      v16bf b = load_mat_b(sB[s] + nt * 16 * 32, 32);
      acc0[nt] = wmma_bf16(a0, b, acc0[nt]);
      acc1[nt] = wmma_bf16(a1, b, acc1[nt]);
    }
    __syncthreads();  // all waves done with buffer s before it is refilled
  }

  // Epilogue: C layout -> lane (N), VGPR-e (M, +8 for upper half-wave)
  const int nlo = lane & 15;
  const int eoff = (lane >> 4) * 8;
#pragma unroll
  for (int nt = 0; nt < 4; ++nt) {
    int col = n0 + nt * 16 + nlo;
#pragma unroll
    for (int e = 0; e < 8; ++e) {
      int row0 = m0 + w * 32 + eoff + e;
      int row1 = row0 + 16;
      float v0 = acc0[nt][e];
      float v1 = acc1[nt][e];
      if (OUTF32) {
        ((float*)Cp)[(size_t)row0 * N + col] = v0 + bias[col];
        ((float*)Cp)[(size_t)row1 * N + col] = v1 + bias[col];
      } else {
        ((__bf16*)Cp)[(size_t)row0 * N + col] = (__bf16)v0;
        ((__bf16*)Cp)[(size_t)row1 * N + col] = (__bf16)v1;
      }
    }
  }
}

// ---------------------------------------------------------------------------
// Kernel 4: flash attention per (64 q-rows, head, batch).
// Block = 128 threads (4 waves); wave owns 16 q rows, full DK=64.
// K tile staged with async global->LDS copies; V staged manually (needs
// transpose); online softmax with bias + key mask; P restaged via LDS.
// ---------------------------------------------------------------------------
__global__ void __launch_bounds__(128)
flash_attn_wmma(const __bf16* __restrict__ Q, const __bf16* __restrict__ Kb,
                const __bf16* __restrict__ Vb, const int* __restrict__ mask,
                const float* __restrict__ biasH, __bf16* __restrict__ O) {
  __shared__ __align__(16) __bf16 sK[64 * 64];        // [key][dk]
  __shared__ __align__(16) __bf16 sVt[64 * 64];       // [dk][key]
  __shared__ __align__(16) __bf16 sP[4 * 16 * 64];    // per-wave [qrow][key]
  __shared__ int sM[64];

  const int t = threadIdx.x;
  const int lane = t & 31;
  const int w = t >> 5;
  const int q0 = blockIdx.x * 64;
  const int h = blockIdx.y;
  const int b = blockIdx.z;
  const int nlo = lane & 15;
  const int eoff = (lane >> 4) * 8;
  const float hb = biasH[h];
  const float inv_scale = 0.125f;      // 1/sqrt(64)
  const size_t rs = (size_t)HH * DKK;  // 1024: row stride in (b,l,h,dk)

  // Q tile (16x64 per wave) preloaded directly from global in A-layout.
  const __bf16* qbase = Q + ((size_t)(b * LL + q0 + w * 16)) * rs + h * DKK;
  v16bf aq0 = load_mat_a(qbase, (int)rs);
  v16bf aq1 = load_mat_a(qbase + 32, (int)rs);

  v8f z = {0.f, 0.f, 0.f, 0.f, 0.f, 0.f, 0.f, 0.f};
  v8f oacc[4];
#pragma unroll
  for (int i = 0; i < 4; ++i) oacc[i] = z;
  float mrun[8], lrun[8];
#pragma unroll
  for (int e = 0; e < 8; ++e) { mrun[e] = -3.0e38f; lrun[e] = 0.0f; }

  for (int kb0 = 0; kb0 < LL; kb0 += 64) {
    __syncthreads();
    // K block: straight copy -> async global->LDS (4 x b128 per thread).
#pragma unroll
    for (int j = 0; j < 4; ++j) {
      int ci = t + j * 128;
      int r = ci >> 3, c = (ci & 7) * 8;
      async_copy_b128(lds_lo32(sK + r * 64 + c),
                      Kb + ((size_t)(b * LL + kb0 + r)) * rs + h * DKK + c);
    }
    // V block: transposed on the way into LDS (manual path).
    for (int i = t; i < 64 * 64 / 4; i += 128) {
      int r = (i * 4) >> 6, c = (i * 4) & 63;
      v4bf vv = *(const v4bf*)(Vb + ((size_t)(b * LL + kb0 + r)) * rs +
                               h * DKK + c);
      sVt[(c + 0) * 64 + r] = vv[0];
      sVt[(c + 1) * 64 + r] = vv[1];
      sVt[(c + 2) * 64 + r] = vv[2];
      sVt[(c + 3) * 64 + r] = vv[3];
    }
    if (t < 64) sM[t] = mask[(size_t)b * LL + kb0 + t];
    WAIT_ASYNC(0x0);
    __syncthreads();

    // S = Q * K^T  (16x64 per wave, 2 k-steps x 4 n-tiles)
    v8f sacc[4];
#pragma unroll
    for (int i = 0; i < 4; ++i) sacc[i] = z;
#pragma unroll
    for (int kk = 0; kk < 64; kk += 32) {
      v16bf a = (kk == 0) ? aq0 : aq1;
#pragma unroll
      for (int nt = 0; nt < 4; ++nt) {
        v16bf bm = load_mat_b(sK + nt * 16 * 64 + kk, 64);
        sacc[nt] = wmma_bf16(a, bm, sacc[nt]);
      }
    }

    // Scale + head bias + key mask; online softmax (rows live per half-wave).
    float s[4][8], rmax[8];
#pragma unroll
    for (int e = 0; e < 8; ++e) rmax[e] = -3.0e38f;
#pragma unroll
    for (int nt = 0; nt < 4; ++nt) {
      int mk = sM[nt * 16 + nlo];
#pragma unroll
      for (int e = 0; e < 8; ++e) {
        float v = sacc[nt][e] * inv_scale + hb;
        if (mk == 0) v = -10000.0f;
        s[nt][e] = v;
        rmax[e] = fmaxf(rmax[e], v);
      }
    }
#pragma unroll
    for (int d = 1; d < 16; d <<= 1)
#pragma unroll
      for (int e = 0; e < 8; ++e)
        rmax[e] = fmaxf(rmax[e], __shfl_xor(rmax[e], d, 32));

    float corr[8], mnew[8];
#pragma unroll
    for (int e = 0; e < 8; ++e) {
      mnew[e] = fmaxf(mrun[e], rmax[e]);
      corr[e] = __expf(mrun[e] - mnew[e]);
      mrun[e] = mnew[e];
    }
    float rsum[8];
#pragma unroll
    for (int e = 0; e < 8; ++e) rsum[e] = 0.0f;
#pragma unroll
    for (int nt = 0; nt < 4; ++nt)
#pragma unroll
      for (int e = 0; e < 8; ++e) {
        float p = __expf(s[nt][e] - mnew[e]);
        s[nt][e] = p;
        rsum[e] += p;
      }
#pragma unroll
    for (int d = 1; d < 16; d <<= 1)
#pragma unroll
      for (int e = 0; e < 8; ++e) rsum[e] += __shfl_xor(rsum[e], d, 32);
#pragma unroll
    for (int e = 0; e < 8; ++e) lrun[e] = lrun[e] * corr[e] + rsum[e];
#pragma unroll
    for (int nt = 0; nt < 4; ++nt)
#pragma unroll
      for (int e = 0; e < 8; ++e) oacc[nt][e] *= corr[e];

    // C-layout P -> per-wave LDS tile [qrow][key] (bf16), then A-layout reads.
#pragma unroll
    for (int nt = 0; nt < 4; ++nt) {
      int keyc = nt * 16 + nlo;
#pragma unroll
      for (int e = 0; e < 8; ++e)
        sP[w * 16 * 64 + (eoff + e) * 64 + keyc] = (__bf16)s[nt][e];
    }
    // Per-wave LDS ordering: stores complete before cross-lane reads.
    asm volatile("s_wait_dscnt 0x0" ::: "memory");

    // O += P * V  (A = P 16x64, B = V^T per dk-tile)
#pragma unroll
    for (int kk = 0; kk < 64; kk += 32) {
      v16bf ap = load_mat_a(sP + w * 16 * 64 + kk, 64);
#pragma unroll
      for (int nt = 0; nt < 4; ++nt) {
        v16bf bv = load_mat_b(sVt + nt * 16 * 64 + kk, 64);
        oacc[nt] = wmma_bf16(ap, bv, oacc[nt]);
      }
    }
  }

  // Normalize and store as bf16 into (b, l, h, dk) for the output projection.
#pragma unroll
  for (int nt = 0; nt < 4; ++nt) {
    int dk = nt * 16 + nlo;
#pragma unroll
    for (int e = 0; e < 8; ++e) {
      int row = q0 + w * 16 + eoff + e;
      float ov = oacc[nt][e] / lrun[e];
      O[((size_t)(b * LL + row)) * rs + h * DKK + dk] = (__bf16)ov;
    }
  }
}

// ---------------------------------------------------------------------------
// Host launcher
// ---------------------------------------------------------------------------
extern "C" void kernel_launch(void* const* d_in, const int* in_sizes, int n_in,
                              void* d_out, int out_size, void* d_ws,
                              size_t ws_size, hipStream_t stream) {
  const float* query = (const float*)d_in[0];
  const float* key   = (const float*)d_in[1];
  const float* value = (const float*)d_in[2];
  const float* Wq    = (const float*)d_in[3];
  const float* Wk    = (const float*)d_in[4];
  const float* Wv    = (const float*)d_in[5];
  const float* cs    = (const float*)d_in[6];
  const float* out_w = (const float*)d_in[7];
  const float* out_b = (const float*)d_in[8];
  const float* nbw   = (const float*)d_in[9];
  const int*   mask  = (const int*)d_in[10];
  const int*   nbidx = (const int*)d_in[11];

  const size_t actN = (size_t)BB * LL * DD;  // 4096 x 1024

  char* ws = (char*)d_ws;
  size_t off = 0;
  float* bias = (float*)(ws + off);           off += 256;
  __bf16* wq_bf = (__bf16*)(ws + off);        off += (size_t)DD * DD * 2;
  __bf16* wk_bf = (__bf16*)(ws + off);        off += (size_t)DD * DD * 2;
  __bf16* wv_bf = (__bf16*)(ws + off);        off += (size_t)DD * DD * 2;
  __bf16* wo_bf = (__bf16*)(ws + off);        off += (size_t)DD * DD * 2;
  __bf16* xq_bf = (__bf16*)(ws + off);        off += actN * 2;
  __bf16* xk_bf = (__bf16*)(ws + off);        off += actN * 2;
  __bf16* xv_bf = (__bf16*)(ws + off);        off += actN * 2;
  __bf16* Qbf = (__bf16*)(ws + off);          off += actN * 2;
  __bf16* Kbf = (__bf16*)(ws + off);          off += actN * 2;
  __bf16* Vbf = (__bf16*)(ws + off);          off += actN * 2;
  __bf16* Abf = (__bf16*)(ws + off);          off += actN * 2;

  // 1) phases -> per-head bias
  phases_bias_kernel<<<1, 32, 0, stream>>>(cs, nbw, nbidx, bias);

  // 2) bf16 conversions: weights + activations (one pass, ~3us of HBM time)
  {
    int nw = DD * DD;
    dim3 cb(256), cgw((unsigned)((nw / 4 + 255) / 256));
    f32_to_bf16_kernel<<<cgw, cb, 0, stream>>>(Wq, wq_bf, nw);
    f32_to_bf16_kernel<<<cgw, cb, 0, stream>>>(Wk, wk_bf, nw);
    f32_to_bf16_kernel<<<cgw, cb, 0, stream>>>(Wv, wv_bf, nw);
    f32_to_bf16_kernel<<<cgw, cb, 0, stream>>>(out_w, wo_bf, nw);
    int na = (int)actN;
    dim3 cga((unsigned)((na / 4 + 255) / 256));
    f32_to_bf16_kernel<<<cga, cb, 0, stream>>>(query, xq_bf, na);
    f32_to_bf16_kernel<<<cga, cb, 0, stream>>>(key, xk_bf, na);
    f32_to_bf16_kernel<<<cga, cb, 0, stream>>>(value, xv_bf, na);
  }

  // 3) QKV projections: (B*L x D) * (H*DK x D)^T -> bf16 (b,l,h,dk)
  {
    dim3 gb(128), gg(DD / 64, (BB * LL) / 128);
    gemm_bt_wmma<false><<<gg, gb, 0, stream>>>(xq_bf, wq_bf, Qbf,
                                               (const float*)nullptr,
                                               BB * LL, DD, DD);
    gemm_bt_wmma<false><<<gg, gb, 0, stream>>>(xk_bf, wk_bf, Kbf,
                                               (const float*)nullptr,
                                               BB * LL, DD, DD);
    gemm_bt_wmma<false><<<gg, gb, 0, stream>>>(xv_bf, wv_bf, Vbf,
                                               (const float*)nullptr,
                                               BB * LL, DD, DD);
  }

  // 4) flash attention
  {
    dim3 ab(128), ag(LL / 64, HH, BB);
    flash_attn_wmma<<<ag, ab, 0, stream>>>(Qbf, Kbf, Vbf, mask, bias, Abf);
  }

  // 5) output projection: (B*L x D) * out_w^T + out_b -> fp32 d_out
  {
    dim3 gb(128), gg(DD / 64, (BB * LL) / 128);
    gemm_bt_wmma<true><<<gg, gb, 0, stream>>>(Abf, wo_bf, d_out, out_b,
                                              BB * LL, DD, DD);
  }
}
